// Model_28174985462416
// MI455X (gfx1250) — matrix operand order repaired and checked by reading, not compiled, not run on hardware
//
#include <hip/hip_runtime.h>

// ---------------------------------------------------------------------------
// Fused windowed-attention transformer layer for MI455X (gfx1250, wave32).
// All GEMMs on v_wmma_f32_16x16x32_f16 (fp32 accumulate), f16 intermediates.
// GEMM uses a double-buffered LDS pipeline (global prefetch overlaps WMMA).
// ---------------------------------------------------------------------------

#define BB 4
#define TT 256
#define NN 16
#define HH 256
#define NH 8
#define DD 32
#define FF 1024
#define WN 8
#define ROWS (BB * TT * NN)   // 16384

typedef __attribute__((ext_vector_type(16))) _Float16 v16h;
typedef __attribute__((ext_vector_type(8)))  _Float16 v8h;
typedef __attribute__((ext_vector_type(8)))  float    v8f;
typedef __attribute__((ext_vector_type(4)))  float    v4f;

static __device__ __forceinline__ v16h cat8(v8h lo, v8h hi) {
    v16h r;
#pragma unroll
    for (int i = 0; i < 8; ++i) { r[i] = lo[i]; r[i + 8] = hi[i]; }
    return r;
}

// 16 contiguous elements -> 16 halfs (f32 source converts, f16 source copies)
static __device__ __forceinline__ void load16h(const float* __restrict__ p, _Float16* h) {
    v4f a = *(const v4f*)p, b = *(const v4f*)(p + 4);
    v4f c = *(const v4f*)(p + 8), d = *(const v4f*)(p + 12);
#pragma unroll
    for (int i = 0; i < 4; ++i) {
        h[i] = (_Float16)a[i]; h[4 + i] = (_Float16)b[i];
        h[8 + i] = (_Float16)c[i]; h[12 + i] = (_Float16)d[i];
    }
}
static __device__ __forceinline__ void load16h(const _Float16* __restrict__ p, _Float16* h) {
    *(v8h*)h       = *(const v8h*)p;
    *(v8h*)(h + 8) = *(const v8h*)(p + 8);
}

static __device__ __forceinline__ int iclamp(int x, int lo, int hi) {
    return x < lo ? lo : (x > hi ? hi : x);
}

// ---------------------------------------------------------------------------
// Generic f16-WMMA GEMM: C(MxN) = act(scale*(A(MxK) @ Bw(KxN) + bias))
// 8 waves, 128x128 block tile, BK=32, each wave a 32x64 tile (2x4 fragments).
// Double-buffered LDS: tile kt+1 is fetched from global into registers while
// tile kt's 8 WMMAs run; one barrier per K-step.
// A tile in LDS as [M][K] (A-fragment = 2x b128), B tile transposed [N][K]
// (B-fragment = 2x b128). Row stride 40 halfs keeps 16B alignment.
// ---------------------------------------------------------------------------
#define TILEH (128 * 40)

template <typename AT>
__global__ __launch_bounds__(256)
void gemm_wmma(const AT* __restrict__ A, const float* __restrict__ Bw,
               const float* __restrict__ bias, _Float16* __restrict__ outH,
               float* __restrict__ outF, int M, int N, int K,
               float scale, int relu)
{
    __shared__ _Float16 As[2 * TILEH];
    __shared__ _Float16 Bs[2 * TILEH];
    const int tid  = threadIdx.x;
    const int lane = tid & 31, wid = tid >> 5;
    const int wm = wid & 3, wn = wid >> 2;
    const int m0 = blockIdx.y * 128, n0 = blockIdx.x * 128;
    const int cl   = lane & 15;
    const int hi16 = lane >> 4;

    v8f acc[2][4];
#pragma unroll
    for (int i = 0; i < 2; ++i)
#pragma unroll
        for (int j = 0; j < 4; ++j) acc[i][j] = (v8f){};

    const int arow = tid >> 1, acb = (tid & 1) << 4;   // 128 rows x 32 k
    const int brow = tid >> 3, bnb = (tid & 7) << 4;   // 32 k x 128 n

    _Float16 ha[16];
    float    fb[16];
    auto fetch = [&](int k0) {
        load16h(A + (size_t)(m0 + arow) * K + k0 + acb, ha);
        const float* bp = Bw + (size_t)(k0 + brow) * N + n0 + bnb;
        v4f x0 = *(const v4f*)bp,       x1 = *(const v4f*)(bp + 4);
        v4f x2 = *(const v4f*)(bp + 8), x3 = *(const v4f*)(bp + 12);
#pragma unroll
        for (int i = 0; i < 4; ++i) {
            fb[i] = x0[i]; fb[4 + i] = x1[i]; fb[8 + i] = x2[i]; fb[12 + i] = x3[i];
        }
    };

    const int nk = K >> 5;
    fetch(0);
    int buf = 0;
    for (int kt = 0; kt < nk; ++kt) {
        _Float16* Asb = As + buf * TILEH;
        _Float16* Bsb = Bs + buf * TILEH;
        *(v8h*)&Asb[arow * 40 + acb]     = *(v8h*)ha;
        *(v8h*)&Asb[arow * 40 + acb + 8] = *(v8h*)(ha + 8);
#pragma unroll
        for (int i = 0; i < 16; ++i) Bsb[(bnb + i) * 40 + brow] = (_Float16)fb[i];
        __syncthreads();

        if (kt + 1 < nk) fetch((kt + 1) << 5);   // overlaps with WMMAs below

        const int kb = hi16 ? 8 : 0;     // A: K pairs 0..7 (+8 in hi half)
        const int ko = hi16 ? 16 : 0;    // B: K pairs 0..7 (+8 in hi half), same as A
        v16h af[2];
#pragma unroll
        for (int fm = 0; fm < 2; ++fm) {
            const int r = (wm * 32 + fm * 16 + cl) * 40;
            af[fm] = cat8(*(const v8h*)&Asb[r + kb], *(const v8h*)&Asb[r + 16 + kb]);
        }
#pragma unroll
        for (int fn = 0; fn < 4; ++fn) {
            const int c = (wn * 64 + fn * 16 + cl) * 40;
            const v16h bf = cat8(*(const v8h*)&Bsb[c + kb], *(const v8h*)&Bsb[c + 16 + kb]);
#pragma unroll
            for (int fm = 0; fm < 2; ++fm)
                acc[fm][fn] = __builtin_amdgcn_wmma_f32_16x16x32_f16(
                    false, af[fm], false, bf, (short)0, acc[fm][fn], false, false);
        }
        buf ^= 1;
    }

    const int hb = hi16 ? 8 : 0;
#pragma unroll
    for (int fm = 0; fm < 2; ++fm)
#pragma unroll
        for (int fn = 0; fn < 4; ++fn) {
            const int col = n0 + wn * 64 + fn * 16 + cl;
            const float bvv = bias[col];
#pragma unroll
            for (int r = 0; r < 8; ++r) {
                const int row = m0 + wm * 32 + fm * 16 + r + hb;
                float val = (acc[fm][fn][r] + bvv) * scale;
                if (relu) val = fmaxf(val, 0.0f);
                if (outH) outH[(size_t)row * N + col] = (_Float16)val;
                if (outF) outF[(size_t)row * N + col] = val;
            }
        }
}

// ---------------------------------------------------------------------------
// Windowed attention: one wave per (b, t, head). q/k/v are f16 (b,t,n, h*32+d),
// so each frame's per-head k slice is a direct-from-global WMMA B fragment.
// Logits (16 dets x 17 frames x 16 keys) kept in 17 C fragments; softmax with
// half-wave shfl_xor reductions; ctx via 9 K-chunks of attn@V from wave-private
// LDS (no barriers needed -- LDS ops are in-order per wave).
// ---------------------------------------------------------------------------
#define ATRS 296                  // attn row stride (halfs), mult. of 8 -> 16B
#define WV_OFF (16 * ATRS)        // 4736 halfs
#define SMW (WV_OFF + 32 * 40)    // + V chunk [32 keys][40] = 6016 halfs/wave

__global__ __launch_bounds__(128)
void attn_win(const _Float16* __restrict__ q, const _Float16* __restrict__ k,
              const _Float16* __restrict__ v, const unsigned char* __restrict__ kpm,
              float* __restrict__ ctx)
{
    __shared__ _Float16 sm[4 * SMW];  // ~47 KB, wave-private slices
    const int lane = threadIdx.x & 31, wid = threadIdx.x >> 5;
    _Float16* attnW = sm + wid * SMW;
    _Float16* vC    = attnW + WV_OFF;
    const int task = blockIdx.x * 4 + wid;      // B*T*NH = 8192 tasks
    const int h  = task & 7;
    const int bt = task >> 3;                   // b*T + t
    const int t  = bt & (TT - 1);
    const int b  = bt >> 8;
    const int cl   = lane & 15;
    const int hi16 = lane >> 4;

    // Q fragment: M = det = cl, K-dim pattern per ISA 16-bit A layout
    const _Float16* qp = q + ((size_t)bt * NN + cl) * HH + h * DD;
    const int qb = hi16 ? 8 : 0;
    const v16h aq = cat8(*(const v8h*)(qp + qb), *(const v8h*)(qp + 16 + qb));
    const int ko = hi16 ? 16 : 0;

    v8f lg[17];
    unsigned mask = 0;
#pragma unroll
    for (int f = 0; f < 17; ++f) {
        const int ft = t + f - WN;
        const int valid = (ft >= 0) & (ft < TT);
        const int fr = iclamp(ft, 0, TT - 1);
        const _Float16* kp = k + ((size_t)((b * TT + fr) * NN + cl)) * HH + h * DD + qb;
        const v16h bk = cat8(*(const v8h*)kp, *(const v8h*)(kp + 16));
        v8f z = (v8f){};
        lg[f] = __builtin_amdgcn_wmma_f32_16x16x32_f16(
            false, aq, false, bk, (short)0, z, false, false);
        const int msk = (!valid) | (int)(kpm[(b * TT + fr) * NN + cl] != 0);
        mask |= (unsigned)(msk != 0) << f;
    }

    // Row softmax across 272 keys (rows live in 16-lane halves; xor<=8 stays inside)
    float mx[8];
#pragma unroll
    for (int r = 0; r < 8; ++r) mx[r] = -__builtin_inff();
#pragma unroll
    for (int f = 0; f < 17; ++f) {
        const bool mk = (mask >> f) & 1u;
#pragma unroll
        for (int r = 0; r < 8; ++r)
            mx[r] = fmaxf(mx[r], mk ? -__builtin_inff() : lg[f][r]);
    }
#pragma unroll
    for (int d = 1; d <= 8; d <<= 1)
#pragma unroll
        for (int r = 0; r < 8; ++r)
            mx[r] = fmaxf(mx[r], __shfl_xor(mx[r], d, 32));

    float sum[8];
#pragma unroll
    for (int r = 0; r < 8; ++r) sum[r] = 0.0f;
#pragma unroll
    for (int f = 0; f < 17; ++f) {
        const bool mk = (mask >> f) & 1u;
#pragma unroll
        for (int r = 0; r < 8; ++r) {
            const float e = mk ? 0.0f : __expf(lg[f][r] - mx[r]);
            lg[f][r] = e;
            sum[r] += e;
        }
    }
#pragma unroll
    for (int d = 1; d <= 8; d <<= 1)
#pragma unroll
        for (int r = 0; r < 8; ++r)
            sum[r] += __shfl_xor(sum[r], d, 32);

    float inv[8];
#pragma unroll
    for (int r = 0; r < 8; ++r) inv[r] = 1.0f / sum[r];

    const int hb = hi16 ? 8 : 0;
#pragma unroll
    for (int f = 0; f < 17; ++f)
#pragma unroll
        for (int r = 0; r < 8; ++r)
            attnW[(r + hb) * ATRS + f * 16 + cl] = (_Float16)(lg[f][r] * inv[r]);
#pragma unroll
    for (int r = 0; r < 8; ++r)               // zero 9th-chunk pad keys 272..287
        attnW[(r + hb) * ATRS + 272 + cl] = (_Float16)0.0f;

    // ctx = attn(16x288) @ V(288x32), 9 K-chunks, V chunk staged per-chunk
    v8f acc0 = (v8f){}, acc1 = (v8f){};
#pragma unroll
    for (int c = 0; c < 9; ++c) {
        const int f  = 2 * c + hi16;                 // lane's key = lane in chunk
        const int fr = iclamp(t + f - WN, 0, TT - 1);
        const _Float16* vp = v + ((size_t)((b * TT + fr) * NN + cl)) * HH + h * DD;
        _Float16* dst = vC + lane * 40;
        *(v8h*)(dst)      = *(const v8h*)(vp);
        *(v8h*)(dst + 8)  = *(const v8h*)(vp + 8);
        *(v8h*)(dst + 16) = *(const v8h*)(vp + 16);
        *(v8h*)(dst + 24) = *(const v8h*)(vp + 24);

        const int kb   = 32 * c + (hi16 ? 16 : 0);
        const int rowb = cl * ATRS;
        const v16h aa = cat8(*(const v8h*)&attnW[rowb + kb],
                             *(const v8h*)&attnW[rowb + kb + 8]);
        v16h b0, b1;
        const int kk = hi16 ? 16 : 0;
#pragma unroll
        for (int j = 0; j < 16; ++j) {
            b0[j] = vC[(kk + j) * 40 + cl];
            b1[j] = vC[(kk + j) * 40 + 16 + cl];
        }
        acc0 = __builtin_amdgcn_wmma_f32_16x16x32_f16(false, aa, false, b0, (short)0, acc0, false, false);
        acc1 = __builtin_amdgcn_wmma_f32_16x16x32_f16(false, aa, false, b1, (short)0, acc1, false, false);
    }

    float* cp = ctx + ((size_t)bt * NN) * HH + h * DD;
#pragma unroll
    for (int r = 0; r < 8; ++r) {
        const int mm = r + hb;
        cp[(size_t)mm * HH + cl]      = acc0[r];
        cp[(size_t)mm * HH + 16 + cl] = acc1[r];
    }
}

// ---------------------------------------------------------------------------
// out = LayerNorm(x + rz) * g + be ; one wave per 256-wide row, 8 rows/block
// ---------------------------------------------------------------------------
__global__ __launch_bounds__(256)
void ln_res(const float* __restrict__ x, const float* __restrict__ rz,
            const float* __restrict__ g, const float* __restrict__ be,
            float* __restrict__ out)
{
    const int lane = threadIdx.x & 31, wid = threadIdx.x >> 5;
    const size_t row  = (size_t)blockIdx.x * 8 + wid;
    const size_t base = row * HH + lane;
    float vv[8];
    float s = 0.0f;
#pragma unroll
    for (int i = 0; i < 8; ++i) {
        vv[i] = x[base + i * 32] + rz[base + i * 32];
        s += vv[i];
    }
#pragma unroll
    for (int d = 1; d <= 16; d <<= 1) s += __shfl_xor(s, d, 32);
    const float mean = s * (1.0f / HH);
    float s2 = 0.0f;
#pragma unroll
    for (int i = 0; i < 8; ++i) { const float dd = vv[i] - mean; s2 += dd * dd; }
#pragma unroll
    for (int d = 1; d <= 16; d <<= 1) s2 += __shfl_xor(s2, d, 32);
    const float rstd = rsqrtf(s2 * (1.0f / HH) + 1e-5f);
#pragma unroll
    for (int i = 0; i < 8; ++i) {
        const int col = lane + i * 32;
        out[base + i * 32] = (vv[i] - mean) * rstd * g[col] + be[col];
    }
}

// ---------------------------------------------------------------------------
extern "C" void kernel_launch(void* const* d_in, const int* in_sizes, int n_in,
                              void* d_out, int out_size, void* d_ws, size_t ws_size,
                              hipStream_t stream)
{
    const float* x            = (const float*)d_in[0];
    const unsigned char* kpm  = (const unsigned char*)d_in[1];
    const float* Wq = (const float*)d_in[2];  const float* bq = (const float*)d_in[3];
    const float* Wk = (const float*)d_in[4];  const float* bk = (const float*)d_in[5];
    const float* Wv = (const float*)d_in[6];  const float* bv = (const float*)d_in[7];
    const float* W1 = (const float*)d_in[8];  const float* b1 = (const float*)d_in[9];
    const float* W2 = (const float*)d_in[10]; const float* b2 = (const float*)d_in[11];
    const float* g1 = (const float*)d_in[12]; const float* be1 = (const float*)d_in[13];
    const float* g2 = (const float*)d_in[14]; const float* be2 = (const float*)d_in[15];
    float* out = (float*)d_out;

    char* ws = (char*)d_ws;
    size_t off = 0;
    auto carve = [&](size_t bytes) -> void* {
        void* p = ws + off;
        off += (bytes + 255) & ~(size_t)255;
        return p;
    };
    _Float16* qf  = (_Float16*)carve((size_t)ROWS * HH * 2);
    _Float16* kf  = (_Float16*)carve((size_t)ROWS * HH * 2);
    _Float16* vf  = (_Float16*)carve((size_t)ROWS * HH * 2);
    float*    cx  = (float*)   carve((size_t)ROWS * HH * 4);   // ctx, reused for FFN2 out
    float*    y   = (float*)   carve((size_t)ROWS * HH * 4);
    _Float16* ff  = (_Float16*)carve((size_t)ROWS * FF * 2);

    const float qscale = 0.17677669529663687f;  // 1/sqrt(32)

    // QKV projections (f16 outputs feed attention WMMA directly)
    gemm_wmma<float><<<dim3(HH / 128, ROWS / 128), 256, 0, stream>>>(
        x, Wq, bq, qf, nullptr, ROWS, HH, HH, qscale, 0);
    gemm_wmma<float><<<dim3(HH / 128, ROWS / 128), 256, 0, stream>>>(
        x, Wk, bk, kf, nullptr, ROWS, HH, HH, 1.0f, 0);
    gemm_wmma<float><<<dim3(HH / 128, ROWS / 128), 256, 0, stream>>>(
        x, Wv, bv, vf, nullptr, ROWS, HH, HH, 1.0f, 0);

    // Windowed attention -> ctx (f32)
    attn_win<<<(BB * TT * NH) / 4, 128, 0, stream>>>(qf, kf, vf, kpm, cx);

    // y = LN(x + ctx)
    ln_res<<<ROWS / 8, 256, 0, stream>>>(x, cx, g1, be1, y);

    // FFN: ff = relu(y@W1 + b1) (f16), ffn2 = ff@W2 + b2 (f32, into cx)
    gemm_wmma<float><<<dim3(FF / 128, ROWS / 128), 256, 0, stream>>>(
        y, W1, b1, ff, nullptr, ROWS, FF, HH, 1.0f, 1);
    gemm_wmma<_Float16><<<dim3(HH / 128, ROWS / 128), 256, 0, stream>>>(
        ff, W2, b2, nullptr, cx, ROWS, HH, FF, 1.0f, 0);

    // out = LN(y + ffn2)
    ln_res<<<ROWS / 8, 256, 0, stream>>>(y, cx, g2, be2, out);
}
